// TransitionUp_84052509982743
// MI455X (gfx1250) — matrix operand order repaired and checked by reading, not compiled, not run on hardware
//
#include <hip/hip_runtime.h>
#include <cstddef>

typedef __attribute__((ext_vector_type(16))) _Float16 v16h;
typedef __attribute__((ext_vector_type(8)))  float    v8f;
typedef __attribute__((ext_vector_type(2)))  _Float16 v2h;
typedef __attribute__((ext_vector_type(4)))  unsigned v4u;   // 16B aligned

#define BN_EPS 1e-5f

// ---------------------------------------------------------------------------
// Fused GEMM + BatchNorm(eval) + ReLU:  Y = relu( (X @ W^T + bias - mu)*s + bet )
// X: [N,K] f32 row-major, W: [O,K] f32 row-major (B = W^T == W rows), Y: [N,O] f32
//
// Block = 256 threads (8 wave32), output tile = 128 rows x 64 cols.
//  - 64 weight cols x full K staged to LDS once as f16 (<= 66.5 KB)
//  - A tile (128x32) ping-pong double buffered: ONE barrier per K-step
//  - all fragment loads via 16B-aligned v4u -> ds_load_b128
//  - 4 v_wmma_f32_16x16x32_f16 per K-step per wave (2x2 register tile)
// ---------------------------------------------------------------------------
__global__ __launch_bounds__(256)
void gemm_bn_relu(const float* __restrict__ X, const float* __restrict__ W,
                  const float* __restrict__ bias,
                  const float* __restrict__ gam, const float* __restrict__ bet,
                  const float* __restrict__ mu,  const float* __restrict__ var,
                  float* __restrict__ Y, int K, int O)
{
    constexpr int AS = 40;                               // A row stride (halves)
    __shared__ alignas(16) _Float16 Alds[2][128 * AS];   // 2 x 10 KB
    __shared__ alignas(16) _Float16 Blds[64 * 520];      // 64 x (K+8)  <= 66.5 KB

    const int tid  = threadIdx.x;
    const int lane = tid & 31;
    const int wave = tid >> 5;
    const int wr   = wave & 3;               // row strip 0..3 (32 rows each)
    const int wc   = wave >> 2;              // col strip 0..1 (32 cols each)
    const int row0 = blockIdx.y * 128;
    const int col0 = blockIdx.x * 64;
    const int BS   = K + 8;                  // B row stride (halves, 16B-mult rows)

    // ---- stage entire B tile (64 output cols x K) once, f32 -> f16 ----
    {
        const int kgrp = K >> 3;             // groups of 8 floats per row
        for (int i = tid; i < 64 * kgrp; i += 256) {
            const int orow = i / kgrp;
            const int kq   = i - orow * kgrp;
            const float4* src = (const float4*)(W + (size_t)(col0 + orow) * K + kq * 8);
            float4 f0 = src[0], f1 = src[1];
            v2h* dst = (v2h*)(Blds + orow * BS + kq * 8);
            v2h q0; q0.x = (_Float16)f0.x; q0.y = (_Float16)f0.y;
            v2h q1; q1.x = (_Float16)f0.z; q1.y = (_Float16)f0.w;
            v2h q2; q2.x = (_Float16)f1.x; q2.y = (_Float16)f1.y;
            v2h q3; q3.x = (_Float16)f1.z; q3.y = (_Float16)f1.w;
            dst[0] = q0; dst[1] = q1; dst[2] = q2; dst[3] = q3;
        }
    }

    const int srow = tid >> 1;               // A staging: 128 rows
    const int skq  = tid & 1;                // A staging: 2 groups of 16 floats
    const float* Abase = X + (size_t)(row0 + srow) * K + skq * 16;

    // helper lambdas -----------------------------------------------------
    auto loadA = [&](int kk, float4 f[4]) {
        const float4* src = (const float4*)(Abase + kk);
        f[0] = src[0]; f[1] = src[1]; f[2] = src[2]; f[3] = src[3];
    };
    auto storeA = [&](int buf, const float4 f[4]) {
        v2h* dst = (v2h*)(&Alds[buf][srow * AS + skq * 16]);
        #pragma unroll
        for (int i = 0; i < 4; ++i) {
            v2h qa; qa.x = (_Float16)f[i].x; qa.y = (_Float16)f[i].y;
            v2h qb; qb.x = (_Float16)f[i].z; qb.y = (_Float16)f[i].w;
            dst[2 * i]     = qa;
            dst[2 * i + 1] = qb;
        }
    };

    v8f acc00 = {}, acc01 = {}, acc10 = {}, acc11 = {};

    const int m   = lane & 15;
    const int hh  = lane >> 4;               // lane half-group
    const int bdw = BS >> 1;                 // B dword stride per col row

    // prologue: stage first A tile
    {
        float4 f[4];
        loadA(0, f);
        storeA(0, f);
    }
    __syncthreads();                         // covers B staging too

    int buf = 0;
    for (int kk = 0; kk < K; kk += 32) {
        // issue next tile's global loads early (hidden under WMMAs)
        float4 fn[4];
        const bool more = (kk + 32) < K;
        if (more) loadA(kk + 32, fn);

        // ---- fragments per ISA 16-bit A(16x32)/B(32x16) stripe tables ----
        union frag { v16h v; v4u q[2]; };
        frag a0, a1, b0, b1;

        const v4u* Ar0 = (const v4u*)(&Alds[buf][(wr * 32 + m) * AS]);
        const v4u* Ar1 = (const v4u*)(&Alds[buf][(wr * 32 + 16 + m) * AS]);
        a0.q[0] = Ar0[hh];     a0.q[1] = Ar0[2 + hh];
        a1.q[0] = Ar1[hh];     a1.q[1] = Ar1[2 + hh];

        const unsigned* BldsD = (const unsigned*)Blds;
        const v4u* Br0 = (const v4u*)(BldsD + (size_t)(wc * 32 + m) * bdw + (kk >> 1));
        const v4u* Br1 = (const v4u*)(BldsD + (size_t)(wc * 32 + 16 + m) * bdw + (kk >> 1));
        b0.q[0] = Br0[hh];     b0.q[1] = Br0[2 + hh];
        b1.q[0] = Br1[hh];     b1.q[1] = Br1[2 + hh];

        acc00 = __builtin_amdgcn_wmma_f32_16x16x32_f16(false, a0.v, false, b0.v,
                                                       (short)0, acc00, false, false);
        acc01 = __builtin_amdgcn_wmma_f32_16x16x32_f16(false, a0.v, false, b1.v,
                                                       (short)0, acc01, false, false);
        acc10 = __builtin_amdgcn_wmma_f32_16x16x32_f16(false, a1.v, false, b0.v,
                                                       (short)0, acc10, false, false);
        acc11 = __builtin_amdgcn_wmma_f32_16x16x32_f16(false, a1.v, false, b1.v,
                                                       (short)0, acc11, false, false);

        if (more) {
            storeA(buf ^ 1, fn);             // safe: nobody reads buf^1 this iter
            __syncthreads();                 // single barrier per K-step
            buf ^= 1;
        }
    }

    // ---- epilogue: BN(eval) + ReLU, store f32 ----
    const int n  = lane & 15;
    const int mg = (lane >> 4) * 8;          // row group within 16x16 tile
    #pragma unroll
    for (int ct = 0; ct < 2; ++ct) {
        const int col = col0 + wc * 32 + ct * 16 + n;
        const float s  = gam[col] * rsqrtf(var[col] + BN_EPS);
        const float bb = bias[col], mm = mu[col], ee = bet[col];
        const v8f& accA = ct ? acc01 : acc00;
        const v8f& accB = ct ? acc11 : acc10;
        #pragma unroll
        for (int r = 0; r < 8; ++r) {
            const int rowA = row0 + wr * 32 + mg + r;
            float yA = (accA[r] + bb - mm) * s + ee;
            Y[(size_t)rowA * O + col] = fmaxf(yA, 0.0f);
        }
        #pragma unroll
        for (int r = 0; r < 8; ++r) {
            const int rowB = row0 + wr * 32 + 16 + mg + r;
            float yB = (accB[r] + bb - mm) * s + ee;
            Y[(size_t)rowB * O + col] = fmaxf(yB, 0.0f);
        }
    }
}

// ---------------------------------------------------------------------------
// Brute-force 3-NN per query within its batch. Coarse points staged in LDS as
// float4 (64 KB) so the hot loop is ONE ds_load_b128 per candidate.
// ---------------------------------------------------------------------------
__global__ __launch_bounds__(256)
void knn3_kernel(const float* __restrict__ p1, const float* __restrict__ p2,
                 int* __restrict__ idx, float* __restrict__ wgt)
{
    __shared__ float4 pts[4096];             // 64 KB

    const int g     = blockIdx.x * 256 + threadIdx.x;   // global query id
    const int batch = g >> 14;                          // 16384 queries per batch
    const float* pb = p2 + (size_t)batch * 4096 * 3;

    for (int j = threadIdx.x; j < 4096; j += 256) {
        float4 v;
        v.x = pb[j * 3 + 0];
        v.y = pb[j * 3 + 1];
        v.z = pb[j * 3 + 2];
        v.w = 0.0f;
        pts[j] = v;
    }
    __syncthreads();

    const float qx = p1[g * 3 + 0];
    const float qy = p1[g * 3 + 1];
    const float qz = p1[g * 3 + 2];

    float d0 = 3.4e38f, d1 = 3.4e38f, d2 = 3.4e38f;
    int   i0 = 0, i1 = 0, i2 = 0;

    #pragma unroll 4
    for (int j = 0; j < 4096; ++j) {
        const float4 p = pts[j];
        const float dx = qx - p.x;
        const float dy = qy - p.y;
        const float dz = qz - p.z;
        const float d  = dx * dx + dy * dy + dz * dz;
        if (d < d2) {
            if (d < d1) {
                d2 = d1; i2 = i1;
                if (d < d0) { d1 = d0; i1 = i0; d0 = d; i0 = j; }
                else        { d1 = d;  i1 = j; }
            } else {
                d2 = d; i2 = j;
            }
        }
    }

    float w0 = 1.0f / (sqrtf(fmaxf(d0, 1e-12f)) + 1e-8f);
    float w1 = 1.0f / (sqrtf(fmaxf(d1, 1e-12f)) + 1e-8f);
    float w2 = 1.0f / (sqrtf(fmaxf(d2, 1e-12f)) + 1e-8f);
    const float inv = 1.0f / (w0 + w1 + w2);

    idx[g * 3 + 0] = batch * 4096 + i0;
    idx[g * 3 + 1] = batch * 4096 + i1;
    idx[g * 3 + 2] = batch * 4096 + i2;
    wgt[g * 3 + 0] = w0 * inv;
    wgt[g * 3 + 1] = w1 * inv;
    wgt[g * 3 + 2] = w2 * inv;
}

// ---------------------------------------------------------------------------
// out[q,:] += w0*h2[i0,:] + w1*h2[i1,:] + w2*h2[i2,:]   (256 ch = 64 float4)
// 64 threads per query, 4 queries per block. h2 (16 MB) is L2-resident.
// ---------------------------------------------------------------------------
__global__ __launch_bounds__(256)
void interp_add_kernel(const float* __restrict__ h2, const int* __restrict__ idx,
                       const float* __restrict__ wgt, float* __restrict__ out)
{
    const int q = blockIdx.x * 4 + (threadIdx.x >> 6);
    const int c = threadIdx.x & 63;

    const int i0 = idx[q * 3 + 0];
    const int i1 = idx[q * 3 + 1];
    const int i2 = idx[q * 3 + 2];
    const float w0 = wgt[q * 3 + 0];
    const float w1 = wgt[q * 3 + 1];
    const float w2 = wgt[q * 3 + 2];

    const float4* f0 = (const float4*)(h2 + (size_t)i0 * 256);
    const float4* f1 = (const float4*)(h2 + (size_t)i1 * 256);
    const float4* f2 = (const float4*)(h2 + (size_t)i2 * 256);
    float4* o = (float4*)(out + (size_t)q * 256);

    float4 a  = o[c];
    float4 v0 = f0[c], v1 = f1[c], v2 = f2[c];
    a.x += w0 * v0.x + w1 * v1.x + w2 * v2.x;
    a.y += w0 * v0.y + w1 * v1.y + w2 * v2.y;
    a.z += w0 * v0.z + w1 * v1.z + w2 * v2.z;
    a.w += w0 * v0.w + w1 * v1.w + w2 * v2.w;
    o[c] = a;
}

// ---------------------------------------------------------------------------
extern "C" void kernel_launch(void* const* d_in, const int* in_sizes, int n_in,
                              void* d_out, int out_size, void* d_ws, size_t ws_size,
                              hipStream_t stream)
{
    const float* p1  = (const float*)d_in[0];   // [65536,3]
    const float* x1  = (const float*)d_in[1];   // [65536,256]
    const float* p2  = (const float*)d_in[2];   // [16384,3]
    const float* x2  = (const float*)d_in[3];   // [16384,512]
    const float* W1  = (const float*)d_in[4];   // [256,256]
    const float* b1  = (const float*)d_in[5];
    const float* g1  = (const float*)d_in[6];
    const float* be1 = (const float*)d_in[7];
    const float* m1  = (const float*)d_in[8];
    const float* v1  = (const float*)d_in[9];
    const float* W2  = (const float*)d_in[10];  // [256,512]
    const float* b2  = (const float*)d_in[11];
    const float* g2  = (const float*)d_in[12];
    const float* be2 = (const float*)d_in[13];
    const float* m2  = (const float*)d_in[14];
    const float* v2  = (const float*)d_in[15];
    // d_in[16], d_in[17]: offsets o1/o2 — fixed sizes, unused.

    float* out = (float*)d_out;                        // [65536,256] = h1 then += up

    // workspace layout
    float* h2  = (float*)d_ws;                         // 16384*256 f32 = 16 MB
    int*   idx = (int*)(h2 + (size_t)16384 * 256);     // 65536*3 int
    float* wgt = (float*)(idx + (size_t)65536 * 3);    // 65536*3 f32

    // h1 = relu(bn(x1 @ W1^T + b1))  ->  d_out
    gemm_bn_relu<<<dim3(256 / 64, 65536 / 128), 256, 0, stream>>>(
        x1, W1, b1, g1, be1, m1, v1, out, /*K=*/256, /*O=*/256);

    // h2 = relu(bn(x2 @ W2^T + b2))  ->  workspace
    gemm_bn_relu<<<dim3(256 / 64, 16384 / 128), 256, 0, stream>>>(
        x2, W2, b2, g2, be2, m2, v2, h2, /*K=*/512, /*O=*/256);

    // 3-NN + weights
    knn3_kernel<<<65536 / 256, 256, 0, stream>>>(p1, p2, idx, wgt);

    // out += weighted interpolation of h2
    interp_add_kernel<<<65536 / 4, 256, 0, stream>>>(h2, idx, wgt, out);
}
